// SwitchLayer_41721312314324
// MI455X (gfx1250) — matrix operand order repaired and checked by reading, not compiled, not run on hardware
//
#include <hip/hip_runtime.h>
#include <hip/hip_bf16.h>

#define N_TOK 131072
#define DIM   512
#define HID   512
#define NEXP  10
#define TILE_M 64
#define MAX_TILES (N_TOK / TILE_M + NEXP)   // upper bound on sum of per-expert tile counts

typedef __attribute__((ext_vector_type(16))) __bf16 v16bf;
typedef __attribute__((ext_vector_type(8)))  float  v8f;

union Frag {
    v16bf v;
    uint4 u[2];
};

__device__ __forceinline__ unsigned short f2bf(float f) {
    unsigned int u = __float_as_uint(f);
    unsigned int r = (u + 0x7FFFu + ((u >> 16) & 1u)) >> 16;   // round-to-nearest-even
    return (unsigned short)r;
}

// ---------------------------------------------------------------- init
__global__ void init_kernel(int* counts) {
    if (threadIdx.x < 16) counts[threadIdx.x] = 0;
}

// ------------------------------------------------- weight fp32 -> bf16, transposed [E,H,D]
__global__ __launch_bounds__(256) void convert_kernel(
    const float* __restrict__ W1, const float* __restrict__ W2,
    unsigned short* __restrict__ W1t, unsigned short* __restrict__ W2t)
{
    const int total = NEXP * DIM * HID;
    int i = blockIdx.x * 256 + threadIdx.x;
    if (i >= total) return;
    int e   = i / (DIM * HID);
    int rem = i - e * DIM * HID;
    int h   = rem / DIM;          // output column (N dim of GEMM)
    int d   = rem - h * DIM;      // K dim
    size_t src = (size_t)e * DIM * HID + (size_t)d * HID + h;
    W1t[i] = f2bf(W1[src]);
    W2t[i] = f2bf(W2[src]);       // D == H so same indexing
}

// ---------------------------------------------------------------- router + argmax
__global__ __launch_bounds__(256) void router_kernel(
    const float* __restrict__ x, const float* __restrict__ Wr,
    const float* __restrict__ br, int* __restrict__ ids,
    int* __restrict__ counts)
{
    __shared__ float wr[NEXP * DIM];   // 20 KB
    for (int i = threadIdx.x; i < NEXP * DIM; i += 256) wr[i] = Wr[i];
    __syncthreads();

    int lane = threadIdx.x & 31;
    int wave = threadIdx.x >> 5;
    int tokBase = blockIdx.x * 32 + wave * 4;

    for (int t = 0; t < 4; ++t) {
        int n = tokBase + t;
        float acc[NEXP];
        #pragma unroll
        for (int e = 0; e < NEXP; ++e) acc[e] = 0.f;
        const float* xp = x + (size_t)n * DIM;
        #pragma unroll
        for (int k = 0; k < DIM / 32; ++k) {
            float xv = xp[lane + 32 * k];
            #pragma unroll
            for (int e = 0; e < NEXP; ++e)
                acc[e] = fmaf(xv, wr[e * DIM + lane + 32 * k], acc[e]);
        }
        float best = -3.0e38f;
        int bi = 0;
        #pragma unroll
        for (int e = 0; e < NEXP; ++e) {
            float v = acc[e];
            v += __shfl_xor(v, 16, 32);
            v += __shfl_xor(v,  8, 32);
            v += __shfl_xor(v,  4, 32);
            v += __shfl_xor(v,  2, 32);
            v += __shfl_xor(v,  1, 32);
            v += br[e];
            if (v > best) { best = v; bi = e; }   // strict > keeps first max (jnp.argmax)
        }
        if (lane == 0) {
            ids[n] = bi;
            atomicAdd(&counts[bi], 1);
        }
    }
}

// ---------------------------------------------------------------- scan
__global__ void scan_kernel(const int* __restrict__ counts,
                            int* __restrict__ offsets,
                            int* __restrict__ tileOff,
                            int* __restrict__ cursor)
{
    if (threadIdx.x == 0) {
        int off = 0, toff = 0;
        for (int e = 0; e < NEXP; ++e) {
            offsets[e] = off;
            cursor[e]  = off;
            tileOff[e] = toff;
            off  += counts[e];
            toff += (counts[e] + TILE_M - 1) / TILE_M;
        }
        offsets[NEXP] = off;
        tileOff[NEXP] = toff;
    }
}

// ---------------------------------------------------------------- gather permutation
__global__ __launch_bounds__(256) void perm_kernel(
    const int* __restrict__ ids, int* __restrict__ cursor, int* __restrict__ perm)
{
    int n = blockIdx.x * 256 + threadIdx.x;
    int e = ids[n];
    int pos = atomicAdd(&cursor[e], 1);
    perm[pos] = n;
}

// ---------------------------------------------------------------- grouped MoE GEMM (bf16 WMMA)
// One block = one 64-row tile of one expert. 8 waves, each wave owns 4 column
// tiles; per column tile it keeps 4 M-tile accumulators so every B-fragment
// load from L2 feeds 4 WMMAs (halves L2 weight traffic vs 32-row tiles).
__global__ __launch_bounds__(256) void moe_gemm_kernel(
    const float* __restrict__ x,
    const unsigned short* __restrict__ W1t,
    const float* __restrict__ b1,
    const unsigned short* __restrict__ W2t,
    const float* __restrict__ b2,
    const int* __restrict__ offsets,
    const int* __restrict__ tileOff,
    const int* __restrict__ perm,
    float* __restrict__ out)
{
    __shared__ unsigned short xs[TILE_M][DIM + 8];   // bf16 x-tile, padded stride (bank-safe)
    __shared__ unsigned short hs[TILE_M][HID + 8];   // bf16 hidden tile

    int t = blockIdx.x;
    if (t >= tileOff[NEXP]) return;                  // uniform early exit

    int e = 0;
    while (tileOff[e + 1] <= t) ++e;
    int rowStart = offsets[e] + (t - tileOff[e]) * TILE_M;
    int rowEnd   = offsets[e + 1];

    // stage x-tile: fp32 gather -> bf16 LDS
    for (int i = threadIdx.x; i < TILE_M * DIM; i += 256) {
        int r = i / DIM, c = i - r * DIM;
        int gr = rowStart + r;
        float v = 0.f;
        if (gr < rowEnd) v = x[(size_t)perm[gr] * DIM + c];
        xs[r][c] = f2bf(v);
    }
    __syncthreads();

    int lane = threadIdx.x & 31;
    int wave = threadIdx.x >> 5;
    int half = lane >> 4;        // 0: lanes 0-15, 1: lanes 16-31
    int l16  = lane & 15;

    // ---------------- GEMM1: hs = relu(xs @ W1[e] + b1[e])
    const unsigned short* Wbase1 = W1t + (size_t)e * HID * DIM;
    #pragma unroll 1
    for (int c = 0; c < 4; ++c) {
        int nCol = (wave * 4 + c) * 16 + l16;                 // B/C column N = lane%16
        const unsigned short* Bb = Wbase1 + (size_t)nCol * DIM;
        v8f acc[4];
        #pragma unroll
        for (int mt = 0; mt < 4; ++mt)
            acc[mt] = (v8f){0.f,0.f,0.f,0.f,0.f,0.f,0.f,0.f};
        #pragma unroll 4
        for (int kt = 0; kt < DIM / 32; ++kt) {
            Frag b;
            const unsigned short* bp = Bb + kt * 32 + half * 8;       // K k0..k0+7, k0+16..k0+23
            b.u[0] = *(const uint4*)bp;
            b.u[1] = *(const uint4*)(bp + 16);
            #pragma unroll
            for (int mt = 0; mt < 4; ++mt) {
                Frag a;
                const unsigned short* ap = &xs[mt * 16 + l16][kt * 32 + half * 8];
                a.u[0] = *(const uint4*)ap;                           // K k0..k0+7
                a.u[1] = *(const uint4*)(ap + 16);                    // K k0+16..k0+23
                acc[mt] = __builtin_amdgcn_wmma_f32_16x16x32_bf16(
                    false, a.v, false, b.v, (short)0, acc[mt], false, false);
            }
        }
        float bias = b1[e * HID + nCol];
        #pragma unroll
        for (int mt = 0; mt < 4; ++mt) {
            #pragma unroll
            for (int r = 0; r < 8; ++r) {         // C: M = mt*16 + r + 8*half
                float v = fmaxf(acc[mt][r] + bias, 0.f);
                hs[mt * 16 + half * 8 + r][nCol] = f2bf(v);
            }
        }
    }
    __syncthreads();

    // ---------------- GEMM2: out = hs @ W2[e] + b2[e], scattered to tokens
    const unsigned short* Wbase2 = W2t + (size_t)e * HID * HID;
    #pragma unroll 1
    for (int c = 0; c < 4; ++c) {
        int nCol = (wave * 4 + c) * 16 + l16;
        const unsigned short* Bb = Wbase2 + (size_t)nCol * HID;
        v8f acc[4];
        #pragma unroll
        for (int mt = 0; mt < 4; ++mt)
            acc[mt] = (v8f){0.f,0.f,0.f,0.f,0.f,0.f,0.f,0.f};
        #pragma unroll 4
        for (int kt = 0; kt < HID / 32; ++kt) {
            Frag b;
            const unsigned short* bp = Bb + kt * 32 + half * 8;
            b.u[0] = *(const uint4*)bp;
            b.u[1] = *(const uint4*)(bp + 16);
            #pragma unroll
            for (int mt = 0; mt < 4; ++mt) {
                Frag a;
                const unsigned short* ap = &hs[mt * 16 + l16][kt * 32 + half * 8];
                a.u[0] = *(const uint4*)ap;
                a.u[1] = *(const uint4*)(ap + 16);
                acc[mt] = __builtin_amdgcn_wmma_f32_16x16x32_bf16(
                    false, a.v, false, b.v, (short)0, acc[mt], false, false);
            }
        }
        float bias = b2[e * HID + nCol];
        #pragma unroll
        for (int mt = 0; mt < 4; ++mt) {
            #pragma unroll
            for (int r = 0; r < 8; ++r) {
                int row = mt * 16 + half * 8 + r;
                int g = rowStart + row;
                if (g < rowEnd) out[(size_t)perm[g] * HID + nCol] = acc[mt][r] + bias;
            }
        }
    }
}

// ---------------------------------------------------------------- launch
extern "C" void kernel_launch(void* const* d_in, const int* in_sizes, int n_in,
                              void* d_out, int out_size, void* d_ws, size_t ws_size,
                              hipStream_t stream) {
    (void)in_sizes; (void)n_in; (void)out_size; (void)ws_size;
    const float* x  = (const float*)d_in[0];
    const float* Wr = (const float*)d_in[1];
    const float* br = (const float*)d_in[2];
    const float* W1 = (const float*)d_in[3];
    const float* b1 = (const float*)d_in[4];
    const float* W2 = (const float*)d_in[5];
    const float* b2 = (const float*)d_in[6];

    float* out = (float*)d_out;
    int*   ids = (int*)d_out + (size_t)N_TOK * HID;   // second tuple output (int32)

    char* ws = (char*)d_ws;
    int* counts  = (int*)(ws + 0);
    int* offsets = (int*)(ws + 64);
    int* tileOff = (int*)(ws + 128);
    int* cursor  = (int*)(ws + 192);
    int* perm    = (int*)(ws + 256);
    unsigned short* W1t = (unsigned short*)(ws + 256 + (size_t)N_TOK * 4);
    unsigned short* W2t = W1t + (size_t)NEXP * DIM * HID;

    init_kernel<<<1, 64, 0, stream>>>(counts);
    convert_kernel<<<(NEXP * DIM * HID) / 256, 256, 0, stream>>>(W1, W2, W1t, W2t);
    router_kernel<<<N_TOK / 32, 256, 0, stream>>>(x, Wr, br, ids, counts);
    scan_kernel<<<1, 32, 0, stream>>>(counts, offsets, tileOff, cursor);
    perm_kernel<<<N_TOK / 256, 256, 0, stream>>>(ids, cursor, perm);
    moe_gemm_kernel<<<MAX_TILES, 256, 0, stream>>>(x, W1t, b1, W2t, b2,
                                                   offsets, tileOff, perm, out);
}
